// GQA_49538152792561
// MI455X (gfx1250) — matrix operand order repaired and checked by reading, not compiled, not run on hardware
//
#include <hip/hip_runtime.h>
#include <hip/hip_bf16.h>

// ---- model dims (match reference) ----
#define S_LEN 4096
#define DIN   1024
#define DIM   1024
#define NH    16
#define NKV   4
#define HD    64
#define KVD   (NKV*HD)   // 256

typedef __attribute__((ext_vector_type(16))) _Float16 v16h;
typedef __attribute__((ext_vector_type(8)))  _Float16 v8h;
typedef __attribute__((ext_vector_type(8)))  float    v8f;
typedef unsigned int uint32;

// Build a 16-half WMMA fragment register group from two 16-byte chunks.
__device__ __forceinline__ v16h make_frag(const _Float16* p0, const _Float16* p1) {
  v8h lo = *(const v8h*)p0;
  v8h hi = *(const v8h*)p1;
  return __builtin_shufflevector(lo, hi, 0,1,2,3,4,5,6,7,8,9,10,11,12,13,14,15);
}
// A-fragment: chunks at +0 and +16 halfs (uniform base + 32-bit lane offset)
__device__ __forceinline__ v16h afrag(const _Float16* base, uint32 off) {
  return make_frag(base + off, base + off + 16);
}
// B-fragment: chunks at +0 and +16 halfs
__device__ __forceinline__ v16h bfrag(const _Float16* base, uint32 off) {
  return make_frag(base + off, base + off + 16);
}

// ---------------- fp32 -> fp16 convert ----------------
__global__ void f32_to_f16_kernel(const float* __restrict__ src,
                                  _Float16* __restrict__ dst, int n) {
  int i = blockIdx.x * blockDim.x + threadIdx.x;
  if (i < n) dst[i] = (_Float16)src[i];
}

// ---------------- WMMA GEMM: C[M,N] = A[M,K] @ W[N,K]^T ----------------
// Per-wave tile: 32 (M) x 64 (N), software-pipelined k loop. Addressing is
// uniform SGPR base (advanced by scalar kb) + loop-invariant 32-bit lane
// offsets (hardware GVS mode) to avoid per-lane u64 adds and spills.
// mode 0: f16 row-major [M,N]; mode 1: f16 transposed [N,M]; mode 2: f32.
__global__ __launch_bounds__(256, 1)
void gemm_wmma(const _Float16* __restrict__ A,
               const _Float16* __restrict__ W,
               _Float16* __restrict__ Ch,
               float* __restrict__ Cf,
               int M, int N, int K, int mode)
{
  int wave = (blockIdx.x * blockDim.x + threadIdx.x) >> 5;
  int lane = threadIdx.x & 31;
  int grp  = lane >> 4;       // half-wave group (0/1)
  int lid  = lane & 15;       // row-in-A / col-in-B,C index
  int nstrips = N >> 6;
  int mb = (wave / nstrips) << 5;   // 32-row M tile
  int nb = (wave % nstrips) << 6;   // 64-col N tile
  if (mb >= M) return;

  // loop-invariant 32-bit element offsets
  uint32 aoff0 = (uint32)((mb + lid) * K + 8*grp);
  uint32 aoff1 = aoff0 + (uint32)(16 * K);
  uint32 woff0 = (uint32)((nb + lid) * K + 8*grp);
  uint32 woff1 = woff0 + (uint32)(16 * K);
  uint32 woff2 = woff0 + (uint32)(32 * K);
  uint32 woff3 = woff0 + (uint32)(48 * K);

  v8f acc[2][4] = {};
  // prologue: fragments for kb = 0
  v16h aC0 = afrag(A, aoff0);
  v16h aC1 = afrag(A, aoff1);
  v16h bC0 = bfrag(W, woff0);
  v16h bC1 = bfrag(W, woff1);
  v16h bC2 = bfrag(W, woff2);
  v16h bC3 = bfrag(W, woff3);

  for (int kb = 32; kb < K; kb += 32) {
    const _Float16* Ak = A + kb;     // scalar base advance
    const _Float16* Wk = W + kb;
    // warm L2 three k-blocks ahead (global_prefetch_b8)
    __builtin_prefetch(Ak + 96 + aoff0, 0, 1);
    __builtin_prefetch(Wk + 96 + woff0, 0, 1);
    // issue next k-step's loads before consuming current fragments
    v16h aN0 = afrag(Ak, aoff0);
    v16h aN1 = afrag(Ak, aoff1);
    v16h bN0 = bfrag(Wk, woff0);
    v16h bN1 = bfrag(Wk, woff1);
    v16h bN2 = bfrag(Wk, woff2);
    v16h bN3 = bfrag(Wk, woff3);
    // 8 wmmas on current fragments (hide the in-flight loads)
    acc[0][0] = __builtin_amdgcn_wmma_f32_16x16x32_f16(false, aC0, false, bC0, (short)0, acc[0][0], false, false);
    acc[0][1] = __builtin_amdgcn_wmma_f32_16x16x32_f16(false, aC0, false, bC1, (short)0, acc[0][1], false, false);
    acc[0][2] = __builtin_amdgcn_wmma_f32_16x16x32_f16(false, aC0, false, bC2, (short)0, acc[0][2], false, false);
    acc[0][3] = __builtin_amdgcn_wmma_f32_16x16x32_f16(false, aC0, false, bC3, (short)0, acc[0][3], false, false);
    acc[1][0] = __builtin_amdgcn_wmma_f32_16x16x32_f16(false, aC1, false, bC0, (short)0, acc[1][0], false, false);
    acc[1][1] = __builtin_amdgcn_wmma_f32_16x16x32_f16(false, aC1, false, bC1, (short)0, acc[1][1], false, false);
    acc[1][2] = __builtin_amdgcn_wmma_f32_16x16x32_f16(false, aC1, false, bC2, (short)0, acc[1][2], false, false);
    acc[1][3] = __builtin_amdgcn_wmma_f32_16x16x32_f16(false, aC1, false, bC3, (short)0, acc[1][3], false, false);
    aC0 = aN0; aC1 = aN1; bC0 = bN0; bC1 = bN1; bC2 = bN2; bC3 = bN3;
  }
  // epilogue k-step
  acc[0][0] = __builtin_amdgcn_wmma_f32_16x16x32_f16(false, aC0, false, bC0, (short)0, acc[0][0], false, false);
  acc[0][1] = __builtin_amdgcn_wmma_f32_16x16x32_f16(false, aC0, false, bC1, (short)0, acc[0][1], false, false);
  acc[0][2] = __builtin_amdgcn_wmma_f32_16x16x32_f16(false, aC0, false, bC2, (short)0, acc[0][2], false, false);
  acc[0][3] = __builtin_amdgcn_wmma_f32_16x16x32_f16(false, aC0, false, bC3, (short)0, acc[0][3], false, false);
  acc[1][0] = __builtin_amdgcn_wmma_f32_16x16x32_f16(false, aC1, false, bC0, (short)0, acc[1][0], false, false);
  acc[1][1] = __builtin_amdgcn_wmma_f32_16x16x32_f16(false, aC1, false, bC1, (short)0, acc[1][1], false, false);
  acc[1][2] = __builtin_amdgcn_wmma_f32_16x16x32_f16(false, aC1, false, bC2, (short)0, acc[1][2], false, false);
  acc[1][3] = __builtin_amdgcn_wmma_f32_16x16x32_f16(false, aC1, false, bC3, (short)0, acc[1][3], false, false);

  #pragma unroll
  for (int mt = 0; mt < 2; ++mt) {
    #pragma unroll
    for (int nt = 0; nt < 4; ++nt) {
      #pragma unroll
      for (int r = 0; r < 8; ++r) {
        int m = mb + 16*mt + r + 8*grp;    // C layout: row = vgpr + 8*(lane/16)
        int n = nb + nt*16 + lid;          //           col = lane%16
        float cv = acc[mt][nt][r];
        if (mode == 0)      Ch[(uint32)(m * N + n)] = (_Float16)cv;
        else if (mode == 1) Ch[(uint32)(n * M + m)] = (_Float16)cv;  // for V
        else                Cf[(uint32)(m * N + n)] = cv;
      }
    }
  }
}

// ---------------- Flash-attention (GQA, causal) ----------------
// One wave per (head h, 16-row query tile). Key loop step = 32.
// Q: [S, DIM] f16 (head h at cols h*64..); K: [S, KVD] f16; Vt: [KVD, S] f16.
__global__ __launch_bounds__(256, 1)
void attn_wmma(const _Float16* __restrict__ Q,
               const _Float16* __restrict__ Kh,
               const _Float16* __restrict__ Vt,
               _Float16* __restrict__ Ctx)
{
  __shared__ _Float16 lds[8][16][32];   // per-wave P staging (C-layout -> A-frag)
  int w    = threadIdx.x >> 5;
  int lane = threadIdx.x & 31;
  int grp  = lane >> 4;
  int lid  = lane & 15;
  int qrow = (blockIdx.x * 8 + w) << 4;
  int h    = blockIdx.y;
  int g    = h >> 2;                    // kv group (GS = 4)
  const float scale = 0.125f;           // 1/sqrt(64)

  // Q A-fragments covering HD=64 (two 16x32 frags)
  uint32 qoff = (uint32)((qrow + lid) * DIM + h * HD + 8*grp);
  v16h qa0 = afrag(Q, qoff);
  v16h qa1 = afrag(Q, qoff + 32);
  // loop-invariant lane offsets for K and Vt fragments
  uint32 koff = (uint32)(lid * KVD + g * HD + 8*grp);
  uint32 voff = (uint32)((g*HD + lid) * S_LEN + 8*grp);

  float rm[8], rs[8];
  #pragma unroll
  for (int r = 0; r < 8; ++r) { rm[r] = -1e30f; rs[r] = 0.f; }
  v8f acc[4] = {};                      // O tile: 16 rows x 64 hd

  int jmax = qrow + 16;                 // causal bound
  for (int j = 0; j < jmax; j += 32) {
    // ---- V B-frags first: independent of scores, overlap with QK^T ----
    const _Float16* Vj = Vt + j;        // scalar base advance
    v16h vb[4];
    #pragma unroll
    for (int t = 0; t < 4; ++t)         // element (k=key, n=hd)
      vb[t] = bfrag(Vj, voff + (uint32)(t*16*S_LEN));
    // ---- scores for 2 key subtiles of 16 ----
    v8f s[2];
    #pragma unroll
    for (int hf = 0; hf < 2; ++hf) {
      const _Float16* Kj = Kh + (j + 16*hf) * KVD;   // scalar base advance
      v16h kb0 = bfrag(Kj, koff);        // hd 0..31
      v16h kb1 = bfrag(Kj, koff + 32);   // hd 32..63
      v8f z = {};
      z = __builtin_amdgcn_wmma_f32_16x16x32_f16(false, qa0, false, kb0, (short)0, z, false, false);
      z = __builtin_amdgcn_wmma_f32_16x16x32_f16(false, qa1, false, kb1, (short)0, z, false, false);
      s[hf] = z;
    }
    // ---- scale + causal mask ----
    #pragma unroll
    for (int hf = 0; hf < 2; ++hf) {
      int kj = j + 16*hf + lid;
      #pragma unroll
      for (int r = 0; r < 8; ++r) {
        int qi = qrow + r + 8*grp;
        float v = s[hf][r] * scale;
        s[hf][r] = (kj > qi) ? -1e30f : v;
      }
    }
    // ---- online softmax (row reductions across 16-lane groups) ----
    #pragma unroll
    for (int r = 0; r < 8; ++r) {
      float mx = fmaxf(s[0][r], s[1][r]);
      #pragma unroll
      for (int off = 1; off < 16; off <<= 1)
        mx = fmaxf(mx, __shfl_xor(mx, off, 32));
      float mnew  = fmaxf(rm[r], mx);
      float alpha = __expf(rm[r] - mnew);
      float p0 = __expf(s[0][r] - mnew);
      float p1 = __expf(s[1][r] - mnew);
      float sum = p0 + p1;
      #pragma unroll
      for (int off = 1; off < 16; off <<= 1)
        sum += __shfl_xor(sum, off, 32);
      rs[r] = rs[r] * alpha + sum;
      rm[r] = mnew;
      #pragma unroll
      for (int t = 0; t < 4; ++t) acc[t][r] *= alpha;
      // stash P (C-layout) into this wave's LDS tile
      lds[w][r + 8*grp][lid]      = (_Float16)p0;
      lds[w][r + 8*grp][16 + lid] = (_Float16)p1;
    }
    // ---- P as A-frag (16x32) from LDS; PV accumulate ----
    const _Float16* pp = &lds[w][lid][0];
    v16h pa = make_frag(pp + 8*grp, pp + 16 + 8*grp);
    #pragma unroll
    for (int t = 0; t < 4; ++t) {
      acc[t] = __builtin_amdgcn_wmma_f32_16x16x32_f16(
          false, pa, false, vb[t], (short)0, acc[t], false, false);
    }
  }
  // ---- normalize + store ctx (f16) ----
  #pragma unroll
  for (int t = 0; t < 4; ++t) {
    #pragma unroll
    for (int r = 0; r < 8; ++r) {
      float o = acc[t][r] / rs[r];
      Ctx[(uint32)((qrow + r + 8*grp) * DIM + h*HD + t*16 + lid)] = (_Float16)o;
    }
  }
}

extern "C" void kernel_launch(void* const* d_in, const int* in_sizes, int n_in,
                              void* d_out, int out_size, void* d_ws, size_t ws_size,
                              hipStream_t stream) {
  const float* x  = (const float*)d_in[0];
  const float* wq = (const float*)d_in[1];
  const float* wk = (const float*)d_in[2];
  const float* wv = (const float*)d_in[3];
  const float* wo = (const float*)d_in[4];

  // workspace layout (f16), ~33.5 MB total
  _Float16* xh  = (_Float16*)d_ws;
  _Float16* wqh = xh  + (size_t)S_LEN * DIN;
  _Float16* wkh = wqh + (size_t)DIM   * DIN;
  _Float16* wvh = wkh + (size_t)KVD   * DIN;
  _Float16* woh = wvh + (size_t)KVD   * DIN;
  _Float16* Qh  = woh + (size_t)DIM   * DIM;
  _Float16* Kh  = Qh  + (size_t)S_LEN * DIM;
  _Float16* Vth = Kh  + (size_t)S_LEN * KVD;
  _Float16* Ctx = Vth + (size_t)KVD   * S_LEN;

  auto cvt = [&](const float* s, _Float16* d, int n) {
    f32_to_f16_kernel<<<(n + 255) / 256, 256, 0, stream>>>(s, d, n);
  };
  cvt(x,  xh,  S_LEN * DIN);
  cvt(wq, wqh, DIM * DIN);
  cvt(wk, wkh, KVD * DIN);
  cvt(wv, wvh, KVD * DIN);
  cvt(wo, woh, DIM * DIM);

  // projections: waves = (M/32)*(N/64), 8 waves (256 threads) per block
  gemm_wmma<<<dim3((S_LEN/32)*(DIM/64)/8), 256, 0, stream>>>(
      xh, wqh, Qh, nullptr, S_LEN, DIM, DIN, 0);
  gemm_wmma<<<dim3((S_LEN/32)*(KVD/64)/8), 256, 0, stream>>>(
      xh, wkh, Kh, nullptr, S_LEN, KVD, DIN, 0);
  gemm_wmma<<<dim3((S_LEN/32)*(KVD/64)/8), 256, 0, stream>>>(
      xh, wvh, Vth, nullptr, S_LEN, KVD, DIN, 1);   // V stored transposed

  // attention: grid.x = qtiles/8, grid.y = heads
  attn_wmma<<<dim3(S_LEN/16/8, NH), 256, 0, stream>>>(Qh, Kh, Vth, Ctx);

  // output projection -> fp32 d_out
  gemm_wmma<<<dim3((S_LEN/32)*(DIM/64)/8), 256, 0, stream>>>(
      Ctx, woh, nullptr, (float*)d_out, S_LEN, DIM, DIM, 2);
}
